// SynthesisLayer_66735201845786
// MI455X (gfx1250) — matrix operand order repaired and checked by reading, not compiled, not run on hardware
//
#include <hip/hip_runtime.h>

typedef __attribute__((ext_vector_type(16))) _Float16 v16h;
typedef __attribute__((ext_vector_type(8)))  _Float16 v8h;
typedef __attribute__((ext_vector_type(8)))  float    v8f;

#define B_      16
#define CIN_    512
#define COUT_   512
#define RES_    64
#define WDIM_   512
#define KK      9

#define NC      66     // cols in LDS x tile (1 halo + 64 + 1 halo)
#define CL      40     // padded cin dim (32 used), 80B stride -> bank-friendly, 16B aligned

// ---------------- K1: styles[b][cin] = w[b]·affine_w[cin]/sqrt(512) + affine_b ----------------
__global__ __launch_bounds__(256) void k_styles(const float* __restrict__ w,
                                                const float* __restrict__ aw,
                                                const float* __restrict__ ab,
                                                float* __restrict__ styles) {
  int g = blockIdx.x * 256 + threadIdx.x;       // 8192 total
  int b = g & 15, cin = g >> 4;
  const float* wr = w  + b   * WDIM_;
  const float* ar = aw + cin * WDIM_;
  float s = 0.f;
  for (int j = 0; j < WDIM_; ++j) s += wr[j] * ar[j];
  styles[b * CIN_ + cin] = s * 0.04419417382415922f + ab[cin];
}

// ---------------- K2: wsq[cout][cin] = sum_kk weight^2 ----------------
__global__ __launch_bounds__(256) void k_wsq(const float* __restrict__ wt,
                                             float* __restrict__ wsq) {
  int g = blockIdx.x * 256 + threadIdx.x;       // 262144 total
  float s = 0.f;
  #pragma unroll
  for (int k = 0; k < KK; ++k) { float v = wt[(size_t)g * KK + k]; s += v * v; }
  wsq[g] = s;
}

// ---------------- K3: demod[b][cout] = rsqrt(sum_cin wsq*styles^2 + eps) ----------------
__global__ __launch_bounds__(256) void k_demod(const float* __restrict__ wsq,
                                               const float* __restrict__ styles,
                                               float* __restrict__ demod) {
  int g = blockIdx.x * 256 + threadIdx.x;       // 8192 total
  int b = g & 15, cout = g >> 4;
  const float* wr = wsq    + (size_t)cout * CIN_;
  const float* sr = styles + b * CIN_;
  float s = 0.f;
  for (int j = 0; j < CIN_; ++j) { float st = sr[j]; s += wr[j] * st * st; }
  demod[b * COUT_ + cout] = rsqrtf(s + 1e-8f);
}

// ---------------- K4: swizzle weights to f16 WMMA A-fragment order ----------------
// layout: [csub(32)][chunk(16)][tap(9)][lane(32)][slot(16)] halfs
// A 16x32 f16 layout: M = lane&15 ; K = slot + (slot>=8?8:0) + (lane>=16?8:0)
__global__ __launch_bounds__(256) void k_wprep(const float* __restrict__ wt,
                                               _Float16* __restrict__ wf16) {
  int csub = blockIdx.x >> 4, chunk = blockIdx.x & 15;
  for (int e = threadIdx.x; e < KK * 512; e += 256) {
    int tap = e >> 9;
    int rem = e & 511;
    int lane = rem >> 4, s = rem & 15;
    int K = s + (s >= 8 ? 8 : 0) + (lane >= 16 ? 8 : 0);
    int cout = csub * 16 + (lane & 15);
    int cin  = chunk * 32 + K;
    wf16[((size_t)(csub * 16 + chunk) * KK + tap) * 512 + lane * 16 + s] =
        (_Float16)wt[((size_t)cout * CIN_ + cin) * KK + tap];
  }
}

// ---------------- K5: xs[b][h][w][cin] = (f16)(x[b][cin][h][w] * styles[b][cin]) ----------------
__global__ __launch_bounds__(256) void k_modx(const float* __restrict__ x,
                                              const float* __restrict__ styles,
                                              _Float16* __restrict__ xs) {
  __shared__ _Float16 ls[32][66];               // 66 cols: bank-conflict-free transpose
  int blk = blockIdx.x;
  int b = blk >> 6, h = blk & 63;
  int t = threadIdx.x;
  int w  = t & 63, c4 = t >> 6;                 // load mapping
  int ci = t & 31, wq = t >> 5;                 // store mapping
  for (int cc = 0; cc < 16; ++cc) {
    int cin0 = cc * 32;
    #pragma unroll
    for (int i = 0; i < 8; ++i) {
      int cin = cin0 + c4 + i * 4;
      ls[c4 + i * 4][w] =
          (_Float16)(x[(((size_t)b * CIN_ + cin) * RES_ + h) * RES_ + w] *
                     styles[b * CIN_ + cin]);
    }
    __syncthreads();
    #pragma unroll
    for (int i = 0; i < 8; ++i) {
      int w2 = wq + i * 8;
      xs[(((size_t)b * RES_ + h) * RES_ + w2) * CIN_ + cin0 + ci] = ls[ci][w2];
    }
    __syncthreads();
  }
}

// ---------------- K6: main conv (implicit GEMM, 9 taps x WMMA f16) ----------------
__global__ __launch_bounds__(256) void k_conv(const _Float16* __restrict__ xs,
                                              const _Float16* __restrict__ wf16,
                                              const float* __restrict__ demod,
                                              const float* __restrict__ bias,
                                              const float* __restrict__ nc,
                                              const float* __restrict__ nsp,
                                              float* __restrict__ out) {
  // double-buffered x tile: [buf][row 0..3 + 1 dummy][col -1..64][cin 0..31 (+pad)]
  __shared__ __align__(16) _Float16 lx[2][5][NC][CL];

  const int t = threadIdx.x;
  const int lane = t & 31, wave = t >> 5;
  const int cb_blk = blockIdx.x * 128;          // cout block
  const int h0 = blockIdx.y * 2;                // 2 output rows per WG
  const int b  = blockIdx.z;

  // zero whole tile once: halo cols + out-of-image rows stay zero forever
  {
    unsigned* p = (unsigned*)&lx[0][0][0][0];
    const int nwords = 2 * 5 * NC * CL / 2;
    for (int i = t; i < nwords; i += 256) p[i] = 0u;
  }
  __syncthreads();

  v8f acc[8];
  #pragma unroll
  for (int i = 0; i < 8; ++i) acc[i] = (v8f)0.0f;

  const int col  = t >> 2;                      // 0..63
  const int part = t & 3;                       // 4 x 16B per 64B cin row

  auto stage = [&](int chunk, int buf) {
    const int cin0 = chunk * 32;
    #pragma unroll
    for (int r = 0; r < 4; ++r) {
      int hh = h0 - 1 + r;
      bool valid = (hh >= 0) && (hh < RES_);
      int lr = valid ? r : 4;                   // dummy row sink keeps async count constant
      int hc = valid ? hh : 0;
      const _Float16* g =
          xs + (((size_t)(b * RES_ + hc) * RES_ + col) * CIN_ + cin0) + part * 8;
      unsigned loff =
          (unsigned)(unsigned long long)(const void*)&lx[buf][lr][col + 1][part * 8];
      asm volatile("global_load_async_to_lds_b128 %0, %1, off"
                   :: "v"(loff), "v"((unsigned long long)(const void*)g)
                   : "memory");
    }
  };

  const int csub = (cb_blk >> 4) + wave;        // this wave's cout16 sub-block
  stage(0, 0);

  for (int chunk = 0; chunk < 16; ++chunk) {
    const int cur = chunk & 1;
    if (chunk + 1 < 16) {
      stage(chunk + 1, cur ^ 1);                // prefetch next chunk (overlaps compute)
      asm volatile("s_wait_asynccnt 4" ::: "memory");   // current chunk's 4 copies done
    } else {
      asm volatile("s_wait_asynccnt 0" ::: "memory");
    }
    __syncthreads();

    const _Float16* wb = wf16 + (size_t)(csub * 16 + chunk) * KK * 512;
    #pragma unroll
    for (int tap = 0; tap < 9; ++tap) {
      const int kh = tap / 3, kw = tap % 3;
      v16h a = *(const v16h*)(wb + tap * 512 + lane * 16);     // A frag: 32B/lane
      #pragma unroll
      for (int nt = 0; nt < 8; ++nt) {
        const int r = (nt >> 2) + kh;                          // LDS row 0..3
        const int c = (nt & 3) * 16 + (lane & 15) + kw;        // LDS col 0..65
        union { v8h h[2]; v16h f; } bv;
        const _Float16* p = &lx[cur][r][c][(lane >> 4) * 8];   // K = s + (s>=8?8:0) + 8*(lane/16)
        bv.h[0] = *(const v8h*)p;
        bv.h[1] = *(const v8h*)(p + 16);
        acc[nt] = __builtin_amdgcn_wmma_f32_16x16x32_f16(
            false, a, false, bv.f, (short)0, acc[nt], false, false);
      }
    }
    __syncthreads();                            // done reading buf before it is re-staged
  }

  // epilogue: demod scale + noise + bias + lrelu*sqrt(2)
  const float nstr = *nsp;
  const int hi = lane >> 4, n = lane & 15;
  float dm[8], bs[8];
  #pragma unroll
  for (int r = 0; r < 8; ++r) {
    int cout = cb_blk + wave * 16 + r + 8 * hi; // C layout: M = r + 8*(lane/16)
    dm[r] = demod[b * COUT_ + cout];
    bs[r] = bias[cout];
  }
  #pragma unroll
  for (int nt = 0; nt < 8; ++nt) {
    int h  = h0 + (nt >> 2);
    int wc = (nt & 3) * 16 + n;                 // C layout: N = lane & 15
    float nz = nc[h * RES_ + wc] * nstr;
    #pragma unroll
    for (int r = 0; r < 8; ++r) {
      int cout = cb_blk + wave * 16 + r + 8 * hi;
      float v = acc[nt][r] * dm[r] + nz + bs[r];
      v = (v >= 0.f ? v : 0.2f * v) * 1.4142135623730951f;
      out[(((size_t)b * COUT_ + cout) * RES_ + h) * RES_ + wc] = v;
    }
  }
}

// ---------------- host ----------------
extern "C" void kernel_launch(void* const* d_in, const int* in_sizes, int n_in,
                              void* d_out, int out_size, void* d_ws, size_t ws_size,
                              hipStream_t stream) {
  const float* x      = (const float*)d_in[0];
  const float* w      = (const float*)d_in[1];
  const float* weight = (const float*)d_in[2];
  const float* bias   = (const float*)d_in[3];
  const float* aw     = (const float*)d_in[4];
  const float* ab     = (const float*)d_in[5];
  const float* nc     = (const float*)d_in[6];
  const float* ns     = (const float*)d_in[7];

  char* ws = (char*)d_ws;
  float*    styles = (float*)(ws + 0);                 //  32 KB
  float*    wsq    = (float*)(ws + 32768);             //   1 MB
  float*    demod  = (float*)(ws + 1081344);           //  32 KB
  _Float16* wf16   = (_Float16*)(ws + 1114112);        // 4.7 MB
  _Float16* xs     = (_Float16*)(ws + 5832704);        //  67 MB

  k_styles<<<32,   256, 0, stream>>>(w, aw, ab, styles);
  k_wsq   <<<1024, 256, 0, stream>>>(weight, wsq);
  k_demod <<<32,   256, 0, stream>>>(wsq, styles, demod);
  k_wprep <<<512,  256, 0, stream>>>(weight, wf16);
  k_modx  <<<1024, 256, 0, stream>>>(x, styles, xs);

  dim3 grid(COUT_ / 128, RES_ / 2, B_);
  k_conv<<<grid, 256, 0, stream>>>(xs, wf16, demod, bias, nc, ns, (float*)d_out);
}
